// EdgeModel_65077344469530
// MI455X (gfx1250) — matrix operand order repaired and checked by reading, not compiled, not run on hardware
//
#include <hip/hip_runtime.h>
#include <hip/hip_bf16.h>

typedef __attribute__((ext_vector_type(16))) __bf16 v16bf;
typedef __attribute__((ext_vector_type(8)))  __bf16 v8bf;
typedef __attribute__((ext_vector_type(4)))  __bf16 v4bf;
typedef __attribute__((ext_vector_type(8)))  float  v8f;
typedef __attribute__((ext_vector_type(4)))  float  v4f;

#define D_NODE 128
#define D_EDGE 64
#define D_GLOB 64
#define D_IN   384
#define HIDDEN 128
#define WPAD   392   // LDS row pitch in bf16: 784B => stride 196 dwords = 4 mod 64 banks
                     // => 16 B-fragment lanes hit 16 distinct bank quartets (conflict-free)
#define NTHREADS 512 // 16 waves/block -> 256 edges/block; amortizes the LDS W fill

// W converted to bf16, row-major [HIDDEN][D_IN]. Rewritten every launch (deterministic).
__device__ __bf16 g_Wb[HIDDEN * D_IN];

__global__ void w_to_bf16_kernel(const float* __restrict__ W, int n4) {
    int i = blockIdx.x * blockDim.x + threadIdx.x;
    if (i < n4) {
        v4f v = ((const v4f*)W)[i];
        v4bf o;
        o[0] = (__bf16)v[0]; o[1] = (__bf16)v[1];
        o[2] = (__bf16)v[2]; o[3] = (__bf16)v[3];
        ((v4bf*)g_Wb)[i] = o;
    }
}

__global__ void f32_to_bf16_vec4(const float* __restrict__ in,
                                 __bf16* __restrict__ out, int n4) {
    int i = blockIdx.x * blockDim.x + threadIdx.x;
    if (i < n4) {
        v4f v = ((const v4f*)in)[i];
        v4bf o;
        o[0] = (__bf16)v[0]; o[1] = (__bf16)v[1];
        o[2] = (__bf16)v[2]; o[3] = (__bf16)v[3];
        ((v4bf*)out)[i] = o;
    }
}

// Block = 512 threads = 16 waves; each wave: one tile of 16 edges x 128 hidden outputs.
// B fragments come from LDS (DScnt), A gathers from global (LOADcnt) -> decoupled waits.
// NB=true: node features pre-converted to bf16 in workspace (halves node gather bytes).
template <bool NB>
__launch_bounds__(NTHREADS)
__global__ void edge_mlp_wmma(const float*  __restrict__ node,
                              const __bf16* __restrict__ nodeb,
                              const float*  __restrict__ edgef,
                              const float*  __restrict__ glob,
                              const int*    __restrict__ eidx,   // [2*E]: row0=src, row1=dst
                              const int*    __restrict__ batch,  // [N]
                              float*        __restrict__ out,    // [E, HIDDEN]
                              int E) {
    __shared__ __bf16 sW[HIDDEN * WPAD];   // ~98 KB of the WGP's 320 KB LDS

    const int tid  = threadIdx.x;
    const int lane = tid & 31;
    const int wave = tid >> 5;
    const int m    = lane & 15;   // edge-row within tile / output column within tile
    const int kh   = lane >> 4;   // K-half selector per WMMA register layout

    // Cooperative LDS fill of bf16 W: 128 rows x 24 chunks of 8 bf16 (16B), coalesced.
    #pragma unroll
    for (int c = 0; c < (HIDDEN * 24) / NTHREADS; ++c) {
        const int idx = c * NTHREADS + tid;
        const int r = idx / 24;
        const int o = idx % 24;
        *(v8bf*)(sW + (size_t)r * WPAD + o * 8) =
            *(const v8bf*)(g_Wb + (size_t)r * D_IN + o * 8);
    }
    __syncthreads();

    const long base = ((long)blockIdx.x * 16 + wave) * 16;
    long e = base + m;
    if (e > (long)E - 1) e = (long)E - 1;   // clamp gathers; stores guarded below

    const int src = eidx[e];
    const int dst = eidx[(size_t)E + e];
    const int g   = batch[src];

    const float*  f0 = node  + (size_t)src * D_NODE;   // cols   0..127 (f32 path)
    const float*  f1 = node  + (size_t)dst * D_NODE;   // cols 128..255 (f32 path)
    const __bf16* b0 = NB ? nodeb + (size_t)src * D_NODE : nullptr;
    const __bf16* b1 = NB ? nodeb + (size_t)dst * D_NODE : nullptr;
    const float*  s2 = edgef + (size_t)e   * D_EDGE;   // cols 256..319
    const float*  s3 = glob  + (size_t)g   * D_GLOB;   // cols 320..383

    v8f acc[8] = {};

    #pragma unroll
    for (int kc = 0; kc < D_IN / 32; ++kc) {
        const int kbase = kc * 32;
        // A-fragment per ISA layout: elems 0-7 <- K = kh*8+0..7, 8-15 <- K = 16+kh*8+0..7
        v16bf a;
        if (NB && kbase < 256) {
            const __bf16* p = (kbase < 128) ? (b0 + kbase) : (b1 + (kbase - 128));
            const v8bf lo = *(const v8bf*)(p + kh * 8);
            const v8bf hi = *(const v8bf*)(p + 16 + kh * 8);
            #pragma unroll
            for (int j = 0; j < 8; ++j) { a[j] = lo[j]; a[8 + j] = hi[j]; }
        } else {
            const float* p;
            if      (kbase < 128) p = f0 + kbase;
            else if (kbase < 256) p = f1 + (kbase - 128);
            else if (kbase < 320) p = s2 + (kbase - 256);
            else                  p = s3 + (kbase - 320);
            const v4f lo0 = *(const v4f*)(p + kh * 8);
            const v4f lo1 = *(const v4f*)(p + kh * 8 + 4);
            const v4f hi0 = *(const v4f*)(p + 16 + kh * 8);
            const v4f hi1 = *(const v4f*)(p + 16 + kh * 8 + 4);
            #pragma unroll
            for (int j = 0; j < 4; ++j) {
                a[j]      = (__bf16)lo0[j];
                a[4 + j]  = (__bf16)lo1[j];
                a[8 + j]  = (__bf16)hi0[j];
                a[12 + j] = (__bf16)hi1[j];
            }
        }

        // B fragments from LDS: lane = output column, elems 0-7 <- K = kh*8+0..7, 8-15 <- K = 16+kh*8+0..7.
        #pragma unroll
        for (int nt = 0; nt < 8; ++nt) {
            const __bf16* bp = sW + (size_t)(nt * 16 + m) * WPAD + kbase;
            const v8bf blo = *(const v8bf*)(bp + kh * 8);
            const v8bf bhi = *(const v8bf*)(bp + 16 + kh * 8);
            v16bf b;
            #pragma unroll
            for (int j = 0; j < 8; ++j) { b[j] = blo[j]; b[8 + j] = bhi[j]; }
            acc[nt] = __builtin_amdgcn_wmma_f32_16x16x32_bf16(
                false, a, false, b, (short)0, acc[nt], false, false);
        }
    }

    // Epilogue: shifted softplus; C/D layout: row = v + kh*8 (VGPR v), col = m.
    #pragma unroll
    for (int nt = 0; nt < 8; ++nt) {
        #pragma unroll
        for (int v = 0; v < 8; ++v) {
            const long row = base + v + kh * 8;
            if (row < (long)E) {
                const float h  = acc[nt][v];
                const float sp = (h > 20.0f) ? h : __logf(1.0f + __expf(h));
                out[(size_t)row * HIDDEN + nt * 16 + m] = sp - 0.69314718055994531f;
            }
        }
    }
}

extern "C" void kernel_launch(void* const* d_in, const int* in_sizes, int n_in,
                              void* d_out, int out_size, void* d_ws, size_t ws_size,
                              hipStream_t stream) {
    const float* node  = (const float*)d_in[0];
    const float* edgef = (const float*)d_in[1];
    const float* glob  = (const float*)d_in[2];
    const int*   eidx  = (const int*)d_in[3];
    const int*   batch = (const int*)d_in[4];
    const float* W     = (const float*)d_in[5];
    float*       out   = (float*)d_out;

    const int E  = in_sizes[3] / 2;
    const int nn = in_sizes[0];          // node_feats element count (N * 128)
    if (E <= 0) return;

    const int wn4 = (HIDDEN * D_IN) / 4;
    w_to_bf16_kernel<<<(wn4 + 255) / 256, 256, 0, stream>>>(W, wn4);

    const bool nb = (ws_size >= (size_t)nn * 2) && (nn % 4 == 0);
    const int epb = 16 * (NTHREADS / 32); // 256 edges per block (16 waves x 16)
    const int blocks = (E + epb - 1) / epb;

    if (nb) {
        __bf16* nodeb = (__bf16*)d_ws;
        const int n4 = nn / 4;
        f32_to_bf16_vec4<<<(n4 + 255) / 256, 256, 0, stream>>>(node, nodeb, n4);
        edge_mlp_wmma<true><<<blocks, NTHREADS, 0, stream>>>(
            node, nodeb, edgef, glob, eidx, batch, out, E);
    } else {
        edge_mlp_wmma<false><<<blocks, NTHREADS, 0, stream>>>(
            node, (const __bf16*)nullptr, edgef, glob, eidx, batch, out, E);
    }
}
